// _PointSAModuleBase_14723147891307
// MI455X (gfx1250) — matrix operand order repaired and checked by reading, not compiled, not run on hardware
//
#include <hip/hip_runtime.h>
#include <hip/hip_bf16.h>
#include <stdint.h>

typedef __attribute__((ext_vector_type(16))) __bf16 v16bf;
typedef __attribute__((ext_vector_type(8)))  __bf16 v8bf;
typedef __attribute__((ext_vector_type(8)))  float  v8f;

#define BB   4
#define NN   16384
#define CIN  64
#define MM   1024
#define KK0  16
#define KK1  32

static const int FEAT_OFF = BB * MM * 3;                 // 12288
static const int IDX_OFF  = FEAT_OFF + BB * 384 * MM;    // 1585152

__device__ __forceinline__ __bf16 f2bf(float f) {
    uint32_t u = __builtin_bit_cast(uint32_t, f);
    u += 0x7FFFu + ((u >> 16) & 1u);                     // round-to-nearest-even
    uint16_t h = (uint16_t)(u >> 16);
    return __builtin_bit_cast(__bf16, h);
}
__device__ __forceinline__ float bf2f(__bf16 b) {
    uint16_t h = __builtin_bit_cast(uint16_t, b);
    uint32_t u = ((uint32_t)h) << 16;
    return __builtin_bit_cast(float, u);
}

// ---------------------------------------------------------------------------
// 1) Farthest point sampling: one 1024-thread workgroup (32 wave32) per batch.
//    dist kept in registers (16 pts/lane). Argmax per iteration:
//      5-step shfl_xor butterfly within each wave (no barriers) ->
//      32 wave winners in LDS -> wave 0 reduces with another butterfly.
//    Only 2 __syncthreads per sequential FPS step.
// ---------------------------------------------------------------------------
__global__ __launch_bounds__(1024) void fps_kernel(const float* __restrict__ xyz,
                                                   float* __restrict__ out) {
    __shared__ float s_v[32];
    __shared__ int   s_i[32];
    __shared__ float s_last[3];
    const int t = threadIdx.x, b = blockIdx.x;
    const int lane = t & 31, wave = t >> 5;
    const float* xb = xyz + (size_t)b * NN * 3;

    float dist[16];
#pragma unroll
    for (int j = 0; j < 16; j++) dist[j] = 1e10f;

    if (t == 0) {
        s_last[0] = xb[0]; s_last[1] = xb[1]; s_last[2] = xb[2];
        size_t o = (size_t)b * MM * 3;
        out[o] = xb[0]; out[o + 1] = xb[1]; out[o + 2] = xb[2];
        ((int*)out)[IDX_OFF + b * MM] = 0;
    }
    __syncthreads();

    for (int it = 1; it < MM; ++it) {
        const float lx = s_last[0], ly = s_last[1], lz = s_last[2];
        float bestv = -1.0f; int besti = 0;
#pragma unroll
        for (int j = 0; j < 16; j++) {
            int p = t + j * 1024;
            float dx = xb[p * 3 + 0] - lx;
            float dy = xb[p * 3 + 1] - ly;
            float dz = xb[p * 3 + 2] - lz;
            float d = dx * dx + dy * dy + dz * dz;
            dist[j] = fminf(dist[j], d);
            if (dist[j] > bestv) { bestv = dist[j]; besti = p; }
        }
        // intra-wave argmax butterfly (wave32)
#pragma unroll
        for (int off = 16; off > 0; off >>= 1) {
            float ov = __shfl_xor(bestv, off, 32);
            int   oi = __shfl_xor(besti, off, 32);
            if (ov > bestv) { bestv = ov; besti = oi; }
        }
        if (lane == 0) { s_v[wave] = bestv; s_i[wave] = besti; }
        __syncthreads();
        if (wave == 0) {
            float v = s_v[lane]; int i2 = s_i[lane];
#pragma unroll
            for (int off = 16; off > 0; off >>= 1) {
                float ov = __shfl_xor(v, off, 32);
                int   oi = __shfl_xor(i2, off, 32);
                if (ov > v) { v = ov; i2 = oi; }
            }
            if (lane == 0) {
                int sel = i2;
                float sx = xb[sel * 3], sy = xb[sel * 3 + 1], sz = xb[sel * 3 + 2];
                s_last[0] = sx; s_last[1] = sy; s_last[2] = sz;
                size_t o = ((size_t)b * MM + it) * 3;
                out[o] = sx; out[o + 1] = sy; out[o + 2] = sz;
                ((int*)out)[IDX_OFF + b * MM + it] = sel;
            }
        }
        __syncthreads();
    }
}

// ---------------------------------------------------------------------------
// 2) Ball query + gather + recenter + concat -> bf16 layer-0 inputs for both
//    scales, laid out [Ncol][Cpad=96] row-major (WMMA B-operand friendly).
//    One thread per query; xyz streamed through LDS chunks; early exit.
// ---------------------------------------------------------------------------
#define CHUNK 2048

__device__ __forceinline__ void emit_row(__bf16* X, size_t row, int Cpad,
                                         float dx, float dy, float dz,
                                         const float* fb, int j) {
    __bf16* r = X + row * (size_t)Cpad;
    r[0] = f2bf(dx); r[1] = f2bf(dy); r[2] = f2bf(dz);
#pragma unroll 4
    for (int c = 0; c < CIN; c++) r[3 + c] = f2bf(fb[(size_t)c * NN + j]);
    for (int c = 3 + CIN; c < Cpad; c++) r[c] = f2bf(0.0f);
}

__global__ __launch_bounds__(128) void ball_gather_kernel(const float* __restrict__ xyz,
                                                          const float* __restrict__ feat,
                                                          const float* __restrict__ out,
                                                          __bf16* __restrict__ X0a,
                                                          __bf16* __restrict__ X0b) {
    __shared__ float sx[CHUNK], sy[CHUNK], sz[CHUNK];
    const int t = threadIdx.x;
    const int bm = blockIdx.x * 128 + t;        // 128 | MM, so b uniform per block
    const int b = bm / MM;
    const float* xb = xyz + (size_t)b * NN * 3;
    const float* fb = feat + (size_t)b * CIN * NN;
    const float qx = out[bm * 3 + 0], qy = out[bm * 3 + 1], qz = out[bm * 3 + 2];
    const float r0sq = 0.2f * 0.2f, r1sq = 0.4f * 0.4f;

    int c0 = 0, c1 = 0, jf0 = -1, jf1 = -1;
    for (int nc = 0; nc < NN; nc += CHUNK) {
        for (int i = t; i < CHUNK; i += 128) {
            int g = nc + i;
            sx[i] = xb[g * 3 + 0]; sy[i] = xb[g * 3 + 1]; sz[i] = xb[g * 3 + 2];
        }
        __syncthreads();
        if (c0 < KK0 || c1 < KK1) {
            for (int i = 0; i < CHUNK; i++) {
                float dx = sx[i] - qx, dy = sy[i] - qy, dz = sz[i] - qz;
                float d2 = dx * dx + dy * dy + dz * dz;
                if (d2 < r1sq) {
                    int j = nc + i;
                    if (d2 < r0sq && c0 < KK0) {
                        if (c0 == 0) jf0 = j;
                        emit_row(X0a, (size_t)bm * KK0 + c0, 96, dx, dy, dz, fb, j);
                        c0++;
                    }
                    if (c1 < KK1) {
                        if (c1 == 0) jf1 = j;
                        emit_row(X0b, (size_t)bm * KK1 + c1, 96, dx, dy, dz, fb, j);
                        c1++;
                    }
                }
                if (c0 >= KK0 && c1 >= KK1) break;
            }
        }
        __syncthreads();
    }
    // fill empty slots with the first found index (0 if none), like CUDA ball_query
    if (c0 < KK0) {
        int j = (jf0 < 0) ? 0 : jf0;
        float dx = xb[j * 3] - qx, dy = xb[j * 3 + 1] - qy, dz = xb[j * 3 + 2] - qz;
        for (int k = c0; k < KK0; k++) emit_row(X0a, (size_t)bm * KK0 + k, 96, dx, dy, dz, fb, j);
    }
    if (c1 < KK1) {
        int j = (jf1 < 0) ? 0 : jf1;
        float dx = xb[j * 3] - qx, dy = xb[j * 3 + 1] - qy, dz = xb[j * 3 + 2] - qz;
        for (int k = c1; k < KK1; k++) emit_row(X0b, (size_t)bm * KK1 + k, 96, dx, dy, dz, fb, j);
    }
}

// ---------------------------------------------------------------------------
// 3) Weight f32 -> bf16 with zero-padding of Cin to a multiple of 32.
// ---------------------------------------------------------------------------
__global__ void convw_kernel(const float* __restrict__ w, __bf16* __restrict__ o,
                             int Cout, int Cin, int Cpad) {
    int e = blockIdx.x * blockDim.x + threadIdx.x;
    if (e >= Cout * Cpad) return;
    int co = e / Cpad, ci = e % Cpad;
    o[e] = (ci < Cin) ? f2bf(w[co * Cin + ci]) : f2bf(0.0f);
}

// ---------------------------------------------------------------------------
// 4) GEMM: Y[Ncol][Cout] (bf16) = X[Ncol][Cpad] (bf16) * W[Cout][Cpad]^T,
//    one 16x16 output tile per wave, K stepped by 32 via
//    v_wmma_f32_16x16x32_bf16. Fragment layouts used here (contiguous k, same for A and B):
//      A lane<16: row=tM+l, K[k0..k0+15] contiguous
//      A lane>=16: row=tM+l, K[k0+16..31] contiguous
//      B lane<16: col=tN+l, K[k0..k0+15] contiguous; lane>=16: K[k0+16..31]
//      D lane: channels tM + half*8 + g, one contiguous 16B bf16 store.
// ---------------------------------------------------------------------------
__global__ __launch_bounds__(256) void gemm_kernel(const __bf16* __restrict__ X,
                                                   const __bf16* __restrict__ W,
                                                   __bf16* __restrict__ Y,
                                                   int Ncol, int Cout, int Cpad) {
    const int wave = threadIdx.x >> 5;
    const int lane = threadIdx.x & 31;
    const int tilesM = Cout >> 4;
    const int tile = blockIdx.x * 8 + wave;
    const int tM = (tile % tilesM) << 4;
    const int tN = (tile / tilesM) << 4;
    if (tN >= Ncol) return;
    const int half = lane >> 4;
    const int l = lane & 15;

    const __bf16* Arow = W + (size_t)(tM + l) * Cpad;
    const __bf16* Brow = X + (size_t)(tN + l) * Cpad;

    v8f acc = {};
    for (int k0 = 0; k0 < Cpad; k0 += 32) {
        v16bf a = *(const v16bf*)(Arow + k0 + half * 16);
        v16bf bfrag = *(const v16bf*)(Brow + k0 + half * 16);
        acc = __builtin_amdgcn_wmma_f32_16x16x32_bf16(
            /*neg_a=*/false, a, /*neg_b=*/false, bfrag,
            /*c_mod=*/(short)0, acc, /*reuse_a=*/false, /*reuse_b=*/false);
    }
    v8bf o;
#pragma unroll
    for (int g = 0; g < 8; g++) o[g] = f2bf(acc[g]);
    *(v8bf*)(Y + (size_t)(tN + l) * Cout + tM + half * 8) = o;
}

// ---------------------------------------------------------------------------
// 5) Per-channel sum / sumsq over Ncol (BN batch stats). LDS ds_add_f32
//    atomics per block, then per-channel global atomics.
// ---------------------------------------------------------------------------
__global__ __launch_bounds__(256) void stats_kernel(const __bf16* __restrict__ Y,
                                                    float* __restrict__ sums,
                                                    int Ncol, int Cout) {
    __shared__ float ls[256], lq[256];
    const int t = threadIdx.x;
    ls[t] = 0.0f; lq[t] = 0.0f;
    __syncthreads();
    const size_t total = (size_t)Ncol * Cout;
    const size_t stride = (size_t)gridDim.x * blockDim.x;
    const int mask = Cout - 1;                       // Cout is a power of two
    for (size_t e = (size_t)blockIdx.x * blockDim.x + t; e < total; e += stride) {
        float v = bf2f(Y[e]);
        int co = (int)(e & mask);
        atomicAdd(&ls[co], v);
        atomicAdd(&lq[co], v * v);
    }
    __syncthreads();
    if (t < Cout) {
        atomicAdd(&sums[t], ls[t]);
        atomicAdd(&sums[256 + t], lq[t]);
    }
}

__global__ void finalize_stats_kernel(const float* __restrict__ sums,
                                      const float* __restrict__ gamma,
                                      const float* __restrict__ beta,
                                      float* __restrict__ ab,
                                      int Cout, float invn) {
    int c = threadIdx.x;
    if (c >= Cout) return;
    float m = sums[c] * invn;
    float v = sums[256 + c] * invn - m * m;
    float inv = rsqrtf(v + 1e-5f);
    float a = gamma[c] * inv;
    ab[c] = a;
    ab[256 + c] = beta[c] - m * a;
}

// ---------------------------------------------------------------------------
// 6) In-place BN + ReLU on a bf16 activation buffer (non-final layers).
// ---------------------------------------------------------------------------
__global__ __launch_bounds__(256) void bn_apply_kernel(__bf16* __restrict__ Y,
                                                       const float* __restrict__ ab,
                                                       size_t total, int Cout) {
    const size_t stride = (size_t)gridDim.x * blockDim.x;
    const int mask = Cout - 1;
    for (size_t e = (size_t)blockIdx.x * blockDim.x + threadIdx.x; e < total; e += stride) {
        int co = (int)(e & mask);
        float v = bf2f(Y[e]);
        Y[e] = f2bf(fmaxf(0.0f, v * ab[co] + ab[256 + co]));
    }
}

// ---------------------------------------------------------------------------
// 7) Final layer: BN + ReLU + max over K, write f32 features (B,384,M).
// ---------------------------------------------------------------------------
__global__ void final_pool_kernel(const __bf16* __restrict__ Y,
                                  const float* __restrict__ ab,
                                  float* __restrict__ out,
                                  int Kk, int Cout, int chbase) {
    const int bm = blockIdx.x;
    const int co = threadIdx.x;
    if (co >= Cout) return;
    const float a = ab[co], bia = ab[256 + co];
    const __bf16* base = Y + (size_t)bm * Kk * Cout + co;
    float mx = -1e30f;
    for (int k = 0; k < Kk; k++) {
        float v = bf2f(base[(size_t)k * Cout]);
        v = fmaxf(0.0f, v * a + bia);
        mx = fmaxf(mx, v);
    }
    const int b = bm / MM, m = bm % MM;
    out[FEAT_OFF + ((size_t)b * 384 + chbase + co) * MM + m] = mx;
}

// ---------------------------------------------------------------------------
// Host orchestration (graph-capture safe: launches + hipMemsetAsync only).
// ---------------------------------------------------------------------------
extern "C" void kernel_launch(void* const* d_in, const int* in_sizes, int n_in,
                              void* d_out, int out_size, void* d_ws, size_t ws_size,
                              hipStream_t stream) {
    (void)in_sizes; (void)n_in; (void)out_size; (void)ws_size;
    const float* xyz  = (const float*)d_in[0];
    const float* feat = (const float*)d_in[1];
    float* out = (float*)d_out;
    char* ws = (char*)d_ws;

    // workspace layout (bytes)
    const size_t XA   = 0;                 // 131072*128*2 = 33,554,432
    const size_t XB   = 33554432;          // same size
    const size_t X1   = 67108864;          // 131072*96*2  = 25,165,824
    const size_t WOFF = 92274688;          // bf16 weights, ~160 KB
    const size_t SUM  = 92434432;          // 512 floats (sum, sumsq)
    const size_t ABO  = 92436480;          // 512 floats (scale, shift)

    __bf16* xa = (__bf16*)(ws + XA);
    __bf16* xb = (__bf16*)(ws + XB);
    __bf16* x1 = (__bf16*)(ws + X1);
    float* sums = (float*)(ws + SUM);
    float* ab   = (float*)(ws + ABO);

    // 1) FPS -> new_xyz + idx into d_out
    fps_kernel<<<BB, 1024, 0, stream>>>(xyz, out);
    // 2) ball query + gather both scales' layer-0 inputs (bf16)
    ball_gather_kernel<<<(BB * MM) / 128, 128, 0, stream>>>(xyz, feat, out, xa, x1);

    // 3) convert weights to padded bf16
    const int cpadL[2][3] = {{96, 64, 64}, {96, 128, 128}};
    const int coutL[2][3] = {{64, 64, 128}, {128, 128, 256}};
    __bf16* wbf[6];
    size_t wacc = 0;
    for (int i = 0; i < 6; i++) {
        int s = i / 3, l = i % 3;
        wbf[i] = (__bf16*)(ws + WOFF) + wacc;
        wacc += (size_t)coutL[s][l] * cpadL[s][l];
    }
    for (int i = 0; i < 6; i++) {
        int s = i / 3, l = i % 3;
        int cout = coutL[s][l], cpad = cpadL[s][l];
        int cin = (l == 0) ? 67 : coutL[s][l - 1];
        int tot = cout * cpad;
        convw_kernel<<<(tot + 255) / 256, 256, 0, stream>>>(
            (const float*)d_in[2 + 3 * i], wbf[i], cout, cin, cpad);
    }

    // 4) per-scale MLP pipeline with bf16 WMMA GEMMs
    for (int s = 0; s < 2; s++) {
        const int Kk   = s ? KK1 : KK0;
        const int Ncol = BB * MM * Kk;
        __bf16* bufA = s ? xa : xb;
        __bf16* bufB = s ? xb : xa;
        __bf16* in   = s ? x1 : xa;
        __bf16* ob   = bufA;
        for (int l = 0; l < 3; l++) {
            const int i = s * 3 + l;
            const int Cout = coutL[s][l], Cpad = cpadL[s][l];
            hipMemsetAsync(sums, 0, 2048, stream);
            const int tiles = (Cout / 16) * (Ncol / 16);
            gemm_kernel<<<tiles / 8, 256, 0, stream>>>(in, wbf[i], ob, Ncol, Cout, Cpad);
            stats_kernel<<<256, 256, 0, stream>>>(ob, sums, Ncol, Cout);
            finalize_stats_kernel<<<1, 256, 0, stream>>>(
                sums, (const float*)d_in[3 + 3 * i], (const float*)d_in[4 + 3 * i],
                ab, Cout, 1.0f / (float)Ncol);
            if (l < 2) {
                bn_apply_kernel<<<1024, 256, 0, stream>>>(ob, ab, (size_t)Ncol * Cout, Cout);
                in = ob;
                ob = (in == bufA) ? bufB : bufA;
            } else {
                final_pool_kernel<<<BB * MM, Cout, 0, stream>>>(ob, ab, out, Kk, Cout, s ? 128 : 0);
            }
        }
    }
}
